// HANEncoder_69389491634561
// MI455X (gfx1250) — matrix operand order repaired and checked by reading, not compiled, not run on hardware
//
#include <hip/hip_runtime.h>
#include <hip/hip_bf16.h>
#include <math.h>

typedef __attribute__((ext_vector_type(16))) _Float16 v16h;
typedef __attribute__((ext_vector_type(8)))  _Float16 v8h;
typedef __attribute__((ext_vector_type(8)))  float    v8f;

// Problem constants (from reference)
#define NC 50000   // nodes
#define EC 400000  // edges per metapath
#define MC 3       // metapaths
#define LC 2       // layers
#define HC 4       // heads
#define DOC 64     // out per head
#define DC 256     // D = H*DO = IN

// ---------------------------------------------------------------------------
// Utility kernels
// ---------------------------------------------------------------------------
__global__ void k_fill(float* __restrict__ p, float v, long n) {
  long i = (long)blockIdx.x * blockDim.x + threadIdx.x;
  if (i < n) p[i] = v;
}

__global__ void k_cvt_f16(const float* __restrict__ in, _Float16* __restrict__ out, long n) {
  long i = (long)blockIdx.x * blockDim.x + threadIdx.x;
  if (i < n) out[i] = (_Float16)in[i];
}

// W is (K x Ncols) row-major fp32 -> Wt is (Ncols x K) row-major f16
__global__ void k_transpose_f16(const float* __restrict__ W, _Float16* __restrict__ Wt,
                                int K, int Ncols) {
  long i = (long)blockIdx.x * blockDim.x + threadIdx.x;
  if (i >= (long)K * Ncols) return;
  int n = (int)(i % Ncols);
  int k = (int)(i / Ncols);
  Wt[(size_t)n * K + k] = (_Float16)W[i];
}

// fp32 transpose (for W1 so the semantic kernel reads contiguously)
__global__ void k_transpose_f32(const float* __restrict__ W, float* __restrict__ Wt,
                                int K, int Ncols) {
  long i = (long)blockIdx.x * blockDim.x + threadIdx.x;
  if (i >= (long)K * Ncols) return;
  int n = (int)(i % Ncols);
  int k = (int)(i / Ncols);
  Wt[(size_t)n * K + k] = W[i];
}

// ---------------------------------------------------------------------------
// WMMA GEMM: C(Mrows x Ncols, f32) = A(Mrows x K, f16) * Bt^T  (+ bias)
// Bt stores B transposed: row n of Bt = column n of B, length K.
// Block = 256 threads = 8 waves; each wave computes a 32x32 tile (2x2 WMMA).
// Grid: x covers 128 cols, y covers 64 rows. Requires K % 32 == 0, Ncols % 32 == 0.
// ---------------------------------------------------------------------------
__global__ __launch_bounds__(256) void k_gemm_wmma(
    const _Float16* __restrict__ A, const _Float16* __restrict__ Bt,
    const float* __restrict__ bias, float* __restrict__ C,
    int Mrows, int K, int Ncols) {
  const int lane = threadIdx.x & 31;
  const int wave = threadIdx.x >> 5;
  const int g    = lane >> 4;   // half-wave group
  const int ln   = lane & 15;
  const int trow = blockIdx.y * 64  + (wave >> 2) * 32;
  const int tcol = blockIdx.x * 128 + (wave & 3) * 32;
  if (tcol >= Ncols) return;  // wave-uniform

  v8f acc[2][2] = {};

#pragma unroll 2
  for (int kb = 0; kb < K; kb += 32) {
    v16h af[2], bf[2];
    // A fragments: 16x32 f16, lane holds row (ln), ks: g*8+e and 16+g*8+e
#pragma unroll
    for (int i = 0; i < 2; i++) {
      int r = trow + i * 16 + ln;
      if (r >= Mrows) r = Mrows - 1;  // clamp (stores are guarded)
      const _Float16* p = A + (size_t)r * K + kb + g * 8;
      v8h lo = *(const v8h*)p;
      v8h hi = *(const v8h*)(p + 16);
#pragma unroll
      for (int t = 0; t < 8; t++) { af[i][t] = lo[t]; af[i][t + 8] = hi[t]; }
    }
    // B fragments: 32x16 f16, lane holds column (ln), ks: g*8+e and 16+g*8+e
#pragma unroll
    for (int j = 0; j < 2; j++) {
      int c = tcol + j * 16 + ln;
      const _Float16* p = Bt + (size_t)c * K + kb + g * 8;
      v8h lo = *(const v8h*)p;
      v8h hi = *(const v8h*)(p + 16);
#pragma unroll
      for (int t = 0; t < 8; t++) { bf[j][t] = lo[t]; bf[j][t + 8] = hi[t]; }
    }
#pragma unroll
    for (int i = 0; i < 2; i++)
#pragma unroll
      for (int j = 0; j < 2; j++)
        acc[i][j] = __builtin_amdgcn_wmma_f32_16x16x32_f16(
            false, af[i], false, bf[j], (short)0, acc[i][j], false, false);
  }

  // C/D layout: VGPR v -> row = trow + i*16 + g*8 + v ; col = tcol + j*16 + ln
  if (trow + 32 <= Mrows) {
    // fast path: whole wave-tile in range, no per-element guards
#pragma unroll
    for (int i = 0; i < 2; i++) {
#pragma unroll
      for (int j = 0; j < 2; j++) {
        int col = tcol + j * 16 + ln;
        float badd = bias ? bias[col] : 0.0f;
        float* cp = C + (size_t)(trow + i * 16 + g * 8) * Ncols + col;
#pragma unroll
        for (int v = 0; v < 8; v++) cp[(size_t)v * Ncols] = acc[i][j][v] + badd;
      }
    }
  } else {
#pragma unroll
    for (int i = 0; i < 2; i++) {
#pragma unroll
      for (int j = 0; j < 2; j++) {
        int col = tcol + j * 16 + ln;
        float badd = bias ? bias[col] : 0.0f;
#pragma unroll
        for (int v = 0; v < 8; v++) {
          int row = trow + i * 16 + g * 8 + v;
          if (row < Mrows) C[(size_t)row * Ncols + col] = acc[i][j][v] + badd;
        }
      }
    }
  }
}

// ---------------------------------------------------------------------------
// Attention scores: el[n,h] = <feat[n,h,:], al[h,:]>, er likewise
// ---------------------------------------------------------------------------
__global__ void k_attn_scores(const float* __restrict__ feat,
                              const float* __restrict__ al, const float* __restrict__ ar,
                              float* __restrict__ el, float* __restrict__ er) {
  long i = (long)blockIdx.x * blockDim.x + threadIdx.x;
  if (i >= (long)NC * HC) return;
  int n = (int)(i >> 2);
  int h = (int)(i & 3);
  const float* f = feat + (size_t)n * DC + h * DOC;
  const float* a = al + h * DOC;
  const float* b = ar + h * DOC;
  float sl = 0.f, sr = 0.f;
  for (int d = 0; d < DOC; d++) { float v = f[d]; sl += v * a[d]; sr += v * b[d]; }
  el[i] = sl; er[i] = sr;
}

__device__ inline void atomicMaxF32(float* addr, float val) {
  unsigned int* ua = (unsigned int*)addr;
  unsigned int old = *ua;
  while (true) {
    float cur = __uint_as_float(old);
    if (cur >= val) break;
    unsigned int assumed = old;
    old = atomicCAS(ua, assumed, __float_as_uint(val));
    if (old == assumed) break;
  }
}

// edge pass 1: e = leaky_relu(el[src]+er[dst]); segment max into mx[dst]
__global__ void k_edge_max(const int* __restrict__ src, const int* __restrict__ dst,
                           const float* __restrict__ el, const float* __restrict__ er,
                           float* __restrict__ eb, float* __restrict__ mx) {
  long i = (long)blockIdx.x * blockDim.x + threadIdx.x;
  if (i >= (long)EC * HC) return;
  int e = (int)(i >> 2);
  int h = (int)(i & 3);
  int s = src[e], d = dst[e];
  float v = el[(size_t)s * HC + h] + er[(size_t)d * HC + h];
  v = v > 0.f ? v : 0.2f * v;
  eb[i] = v;
  atomicMaxF32(&mx[(size_t)d * HC + h], v);
}

// edge pass 2: ex = exp(e - mx[dst]); denom[dst] += ex
__global__ void k_edge_exp(const int* __restrict__ dst, float* __restrict__ eb,
                           const float* __restrict__ mx, float* __restrict__ dn) {
  long i = (long)blockIdx.x * blockDim.x + threadIdx.x;
  if (i >= (long)EC * HC) return;
  int e = (int)(i >> 2);
  int h = (int)(i & 3);
  int d = dst[e];
  float m = mx[(size_t)d * HC + h];
  if (!(m > -1e30f)) m = 0.f;
  float ex = __expf(eb[i] - m);
  eb[i] = ex;
  atomicAdd(&dn[(size_t)d * HC + h], ex);
}

// edge pass 3: msg[dst, c] += ex[e, h(c)] * feat[src, c]
// One thread handles 4 consecutive channels: 1 float4 load + 4 f32 atomics.
__global__ void k_edge_msg(const int* __restrict__ src, const int* __restrict__ dst,
                           const float* __restrict__ eb, const float* __restrict__ feat,
                           float* __restrict__ msg) {
  long i = (long)blockIdx.x * blockDim.x + threadIdx.x;
  if (i >= (long)EC * (DC / 4)) return;
  int e  = (int)(i >> 6);
  int c4 = (int)(i & 63) << 2;     // channel base, multiple of 4 (same head)
  int h  = c4 >> 6;
  int s = src[e], d = dst[e];
  float w = eb[(size_t)e * HC + h];
  const float4 f = *(const float4*)(feat + (size_t)s * DC + c4);
  float* mp = msg + (size_t)d * DC + c4;
  atomicAdd(mp + 0, w * f.x);
  atomicAdd(mp + 1, w * f.y);
  atomicAdd(mp + 2, w * f.z);
  atomicAdd(mp + 3, w * f.w);
}

// finalize: h_out = elu(msg/denom + bias)
__global__ void k_gat_finalize(float* __restrict__ hout, const float* __restrict__ dn,
                               const float* __restrict__ bias) {
  long i = (long)blockIdx.x * blockDim.x + threadIdx.x;
  if (i >= (long)NC * DC) return;
  int n = (int)(i >> 8);
  int c = (int)(i & 255);
  int h = c >> 6;
  float d = fmaxf(dn[(size_t)n * HC + h], 1e-9f);
  float v = hout[i] / d + bias[c];
  hout[i] = v > 0.f ? v : (__expf(v) - 1.0f);
}

// semantic attention score: per (node, metapath) scalar, summed into wsum[m]
// z layout here: (M, N, D); W1t is transposed (DO x D) so reads are contiguous
__global__ void k_semantic(const float* __restrict__ z, const float* __restrict__ W1t,
                           const float* __restrict__ b1, const float* __restrict__ W2,
                           float* __restrict__ wsum) {
  long i = (long)blockIdx.x * blockDim.x + threadIdx.x;
  if (i >= (long)NC * MC) return;
  int n = (int)(i / MC);
  int m = (int)(i % MC);
  const float* zr = z + ((size_t)m * NC + n) * DC;
  float s = 0.f;
  for (int j = 0; j < DOC; j++) {
    const float* w = W1t + (size_t)j * DC;
    float t = b1[j];
    for (int k = 0; k < DC; k++) t += zr[k] * w[k];
    s += tanhf(t) * W2[j];
  }
  atomicAdd(&wsum[m], s);
}

__global__ void k_softmax_beta(const float* __restrict__ wsum, float* __restrict__ beta,
                               float invN) {
  if (threadIdx.x == 0) {
    float w0 = wsum[0] * invN, w1 = wsum[1] * invN, w2 = wsum[2] * invN;
    float mx = fmaxf(w0, fmaxf(w1, w2));
    float e0 = __expf(w0 - mx), e1 = __expf(w1 - mx), e2 = __expf(w2 - mx);
    float s = e0 + e1 + e2;
    beta[0] = e0 / s; beta[1] = e1 / s; beta[2] = e2 / s;
  }
}

// agg[n,c] = sum_m beta[m] * z[m,n,c]
__global__ void k_aggregate(const float* __restrict__ z, const float* __restrict__ beta,
                            float* __restrict__ agg) {
  long i = (long)blockIdx.x * blockDim.x + threadIdx.x;
  if (i >= (long)NC * DC) return;
  float b0 = beta[0], b1 = beta[1], b2 = beta[2];
  const size_t S = (size_t)NC * DC;
  agg[i] = b0 * z[i] + b1 * z[S + i] + b2 * z[2 * S + i];
}

// ---------------------------------------------------------------------------
// Host launch
// ---------------------------------------------------------------------------
static inline unsigned nblk(long n) { return (unsigned)((n + 255) / 256); }

extern "C" void kernel_launch(void* const* d_in, const int* in_sizes, int n_in,
                              void* d_out, int out_size, void* d_ws, size_t ws_size,
                              hipStream_t stream) {
  (void)in_sizes; (void)n_in; (void)out_size; (void)ws_size;
  const float* x        = (const float*)d_in[0];   // (M,N,IN)
  const float* W_gat    = (const float*)d_in[1];   // (L,M,IN,D)
  const float* attn_l   = (const float*)d_in[2];   // (L,M,H,DO)
  const float* attn_r   = (const float*)d_in[3];
  const float* bias_gat = (const float*)d_in[4];   // (L,M,D)
  const float* W1       = (const float*)d_in[5];   // (D,DO)
  const float* b1       = (const float*)d_in[6];
  const float* W2       = (const float*)d_in[7];   // (DO,1)
  const float* Wp       = (const float*)d_in[8];   // (D,DO)
  const float* bp       = (const float*)d_in[9];
  const int*   src      = (const int*)d_in[10];    // (M,E)
  const int*   dst      = (const int*)d_in[11];

  // Workspace carve-out
  size_t off = 0;
  char* base = (char*)d_ws;
  auto take = [&](size_t bytes) -> char* {
    char* p = base + off;
    off += (bytes + 255) & ~(size_t)255;
    return p;
  };
  float*    hA   = (float*)take((size_t)MC * NC * DC * 4);  // layer-0 outputs
  float*    hB   = (float*)take((size_t)MC * NC * DC * 4);  // layer-1 outputs (= z)
  float*    feat = (float*)take((size_t)NC * DC * 4);       // per-conv feat / agg
  _Float16* h16  = (_Float16*)take((size_t)NC * DC * 2);    // f16 activations
  _Float16* w16  = (_Float16*)take((size_t)DC * DC * 2);    // f16 transposed weights
  float*    el   = (float*)take((size_t)NC * HC * 4);
  float*    er   = (float*)take((size_t)NC * HC * 4);
  float*    mx   = (float*)take((size_t)NC * HC * 4);
  float*    dn   = (float*)take((size_t)NC * HC * 4);
  float*    eb   = (float*)take((size_t)EC * HC * 4);       // per-edge scores/exp
  float*    W1t  = (float*)take((size_t)DOC * DC * 4);      // transposed W1 (fp32)
  float*    wsum = (float*)take(64);
  float*    beta = (float*)take(64);

  const long nND = (long)NC * DC;
  const long nNH = (long)NC * HC;
  const long nEH = (long)EC * HC;
  const long nE4 = (long)EC * (DC / 4);

  for (int l = 0; l < LC; l++) {
    float* hcur = (l == 0) ? hA : hB;
    for (int m = 0; m < MC; m++) {
      const float* hin = (l == 0) ? (x + (size_t)m * NC * DC)
                                  : (hA + (size_t)m * NC * DC);
      float* hout = hcur + (size_t)m * NC * DC;

      // Dense part: feat = hin @ W_gat[l][m]  (WMMA f16->f32)
      k_cvt_f16<<<nblk(nND), 256, 0, stream>>>(hin, h16, nND);
      const float* Wlm = W_gat + ((size_t)l * MC + m) * DC * DC;
      k_transpose_f16<<<nblk((long)DC * DC), 256, 0, stream>>>(Wlm, w16, DC, DC);
      dim3 gg(2, (NC + 63) / 64);  // 256 cols / 128 per block
      k_gemm_wmma<<<gg, 256, 0, stream>>>(h16, w16, nullptr, feat, NC, DC, DC);

      // Attention scores
      const float* al = attn_l + ((size_t)l * MC + m) * HC * DOC;
      const float* ar = attn_r + ((size_t)l * MC + m) * HC * DOC;
      k_attn_scores<<<nblk(nNH), 256, 0, stream>>>(feat, al, ar, el, er);

      // Edge softmax + message passing (atomics resolve in 192MB L2)
      k_fill<<<nblk(nNH), 256, 0, stream>>>(mx, -INFINITY, nNH);
      k_fill<<<nblk(nNH), 256, 0, stream>>>(dn, 0.0f, nNH);
      k_fill<<<nblk(nND), 256, 0, stream>>>(hout, 0.0f, nND);
      const int* sm = src + (size_t)m * EC;
      const int* dm = dst + (size_t)m * EC;
      k_edge_max<<<nblk(nEH), 256, 0, stream>>>(sm, dm, el, er, eb, mx);
      k_edge_exp<<<nblk(nEH), 256, 0, stream>>>(dm, eb, mx, dn);
      k_edge_msg<<<nblk(nE4), 256, 0, stream>>>(sm, dm, eb, feat, hout);

      const float* bg = bias_gat + ((size_t)l * MC + m) * DC;
      k_gat_finalize<<<nblk(nND), 256, 0, stream>>>(hout, dn, bg);
    }
  }

  // Semantic attention over metapaths (z = hB, layout (M,N,D))
  k_transpose_f32<<<nblk((long)DC * DOC), 256, 0, stream>>>(W1, W1t, DC, DOC);
  k_fill<<<1, 256, 0, stream>>>(wsum, 0.0f, 8);
  k_semantic<<<nblk((long)NC * MC), 256, 0, stream>>>(hB, W1t, b1, W2, wsum);
  k_softmax_beta<<<1, 32, 0, stream>>>(wsum, beta, 1.0f / (float)NC);
  k_aggregate<<<nblk(nND), 256, 0, stream>>>(hB, beta, feat);

  // Final projection: out = agg @ Wp + bp  (WMMA, Ncols=64)
  k_cvt_f16<<<nblk(nND), 256, 0, stream>>>(feat, h16, nND);
  k_transpose_f16<<<nblk((long)DC * DOC), 256, 0, stream>>>(Wp, w16, DC, DOC);
  dim3 gp(1, (NC + 63) / 64);
  k_gemm_wmma<<<gp, 256, 0, stream>>>(h16, w16, bp, (float*)d_out, NC, DC, DOC);
}
